// _Regression_Layer_24550033064061
// MI455X (gfx1250) — matrix operand order repaired and checked by reading, not compiled, not run on hardware
//
#include <hip/hip_runtime.h>
#include <hip/hip_bf16.h>
#include <cstdint>

// ---------------------------------------------------------------------------
// Problem constants (from the reference)
// ---------------------------------------------------------------------------
#define RN   256          // number of rois
#define RT   8            // T
#define RP   7            // P
#define RC   256          // C
#define RB   2            // batch
#define SAMP 49           // P*P
#define ROWS 100352       // RN*RT*SAMP  (rows of pooled / K of big gemm)
#define D1   2048
#define D2   512
#define DOUT 32           // T*4
#define KSPLIT 8
#define KCHUNK (ROWS / KSPLIT)   // 12544

// inv-norm table sizes per level
#define INV_L2_OFF 0
#define INV_L3_OFF 50176      // 2*8*56*56
#define INV_L4_OFF 62720      // +2*8*28*28
#define INV_L5_OFF 65856      // +2*8*14*14
#define INV_TOTAL  66640      // +2*8*7*7

// workspace byte offsets
#define O_INVN   0ull
#define O_POOLED 266752ull                         // invn padded
#define O_YT     (O_POOLED + 51380224ull)          // pooled f16: ROWS*RC*2
#define O_H1P    (O_YT + 51380224ull)              // yT f16: RN*ROWS*2
#define O_H1     (O_H1P + 16777216ull)             // partials: KSPLIT*RN*D1*4
#define O_H2     (O_H1 + 2097152ull)               // h1 f32: RN*D1*4

typedef _Float16 v16h __attribute__((ext_vector_type(16)));
typedef _Float16 v8h  __attribute__((ext_vector_type(8)));
typedef _Float16 v4h  __attribute__((ext_vector_type(4)));
typedef _Float16 v2h  __attribute__((ext_vector_type(2)));
typedef float    v8f  __attribute__((ext_vector_type(8)));
typedef float    v4f  __attribute__((ext_vector_type(4)));
typedef unsigned int u32x4 __attribute__((ext_vector_type(4)));
typedef int          i32x4 __attribute__((ext_vector_type(4)));
typedef int          i32x8 __attribute__((ext_vector_type(8)));

#define LDS_PITCH 40   // halves; 80B rows: 16 DWORDs data + 4 DWORDs pad

#if defined(__has_builtin)
# if __has_builtin(__builtin_amdgcn_tensor_load_to_lds) && \
     __has_builtin(__builtin_amdgcn_s_wait_tensorcnt)
#  define HAVE_TDM 1
# endif
#endif

#ifdef HAVE_TDM
// ---------------------------------------------------------------------------
// TDM: DMA a [rows x 32] f16 tile (row stride = strideElems) into LDS with a
// 4-DWORD pad every 16 DWORDs (== LDS_PITCH row layout).  D# per ISA 8.3-8.5.
// ---------------------------------------------------------------------------
__device__ __forceinline__ void tdm_load_tile_f16(const _Float16* g, long strideElems,
                                                  unsigned ldsByteOff, int rows) {
  unsigned long long ga = (unsigned long long)(uintptr_t)g;
  const unsigned TD0 = 0x10000000u;              // huge tensor_dim0: never clips
  u32x4 g0;
  g0[0] = 1u;                                    // count=1, is_restore=0
  g0[1] = ldsByteOff;                            // lds_addr
  g0[2] = (unsigned)ga;                          // global_addr[31:0]
  g0[3] = (unsigned)((ga >> 32) & 0x1FFFFFFull) | (2u << 30);  // addr[56:32], type=2
  unsigned long long s0 = (unsigned long long)strideElems;
  i32x8 g1;
  g1[0] = (1 << 16) | (1 << 20) | (3 << 22) | (3 << 25);  // 2B elem, pad 4DW/16DW
  g1[1] = (int)((TD0 & 0xFFFFu) << 16);                   // tensor_dim0[15:0]
  g1[2] = (int)((TD0 >> 16) | ((unsigned)rows << 16));    // td0[31:16] | td1[15:0]
  g1[3] = (int)((((unsigned)rows >> 16) & 0xFFFFu) | (32u << 16)); // td1 hi | tile_dim0=32
  g1[4] = rows;                                           // tile_dim1 (tile_dim2=0)
  g1[5] = (int)(unsigned)s0;                              // tensor_dim0_stride lo32
  g1[6] = (int)(unsigned)(s0 >> 32);                      // stride hi16 | dim1_stride lo16
  g1[7] = 0;
  i32x4 z4 = {0, 0, 0, 0};
#if __clang_major__ >= 23
  i32x8 z8 = {0, 0, 0, 0, 0, 0, 0, 0};
  __builtin_amdgcn_tensor_load_to_lds(g0, g1, z4, z4, z8, 0);
#else
  __builtin_amdgcn_tensor_load_to_lds(g0, g1, z4, z4, 0);
#endif
}
#endif

// ---------------------------------------------------------------------------
// WMMA fragment loaders (CDNA5 16-bit A 16x32 / B 32x16 lane layouts)
// ---------------------------------------------------------------------------
__device__ __forceinline__ v16h frag_a(const _Float16* lds, int rowBase, int lane) {
  int r  = rowBase + (lane & 15);
  int k0 = (lane & 16) ? 8 : 0;          // lanes 16-31 hold K=8..15 / 24..31
  v8h lo = *(const v8h*)(lds + r * LDS_PITCH + k0);
  v8h hi = *(const v8h*)(lds + r * LDS_PITCH + k0 + 16);
  return __builtin_shufflevector(lo, hi, 0,1,2,3,4,5,6,7,8,9,10,11,12,13,14,15);
}

__device__ __forceinline__ v16h frag_b(const _Float16* lds, int colBase, int lane) {
  int r  = colBase + (lane & 15);        // lds holds B transposed: [N][K]
  int k0 = (lane & 16) ? 8 : 0;          // lanes 16-31 hold K=8..15 / 24..31
  v8h lo = *(const v8h*)(lds + r * LDS_PITCH + k0);
  v8h hi = *(const v8h*)(lds + r * LDS_PITCH + k0 + 16);
  return __builtin_shufflevector(lo, hi, 0,1,2,3,4,5,6,7,8,9,10,11,12,13,14,15);
}

#define WMMA_F16(a, b, c) \
  __builtin_amdgcn_wmma_f32_16x16x32_f16(false, (a), false, (b), (short)0, (c), false, false)

// Stage a [rows x 32] f16 tile (A-style, row-major) into LDS: TDM if available.
#define STAGE_A_TILE(lA_, gptr_, strideElems_, tid_)                          \
  do {                                                                        \
    if ((tid_) == 0)                                                          \
      tdm_load_tile_f16((gptr_), (strideElems_),                              \
                        (unsigned)(uintptr_t)&(lA_)[0], 256);                 \
  } while (0)

// ---------------------------------------------------------------------------
// 1) inverse channel L2-norm per (level, b, t, y, x)
// ---------------------------------------------------------------------------
__global__ __launch_bounds__(256) void norm_kernel(const float* __restrict__ f2,
                                                   const float* __restrict__ f3,
                                                   const float* __restrict__ f4,
                                                   const float* __restrict__ f5,
                                                   float* __restrict__ invn) {
  int p = blockIdx.x * 256 + threadIdx.x;
  if (p >= INV_TOTAL) return;
  const float* f; int HW; int local;
  if (p < INV_L3_OFF)      { f = f2; HW = 3136; local = p; }
  else if (p < INV_L4_OFF) { f = f3; HW = 784;  local = p - INV_L3_OFF; }
  else if (p < INV_L5_OFF) { f = f4; HW = 196;  local = p - INV_L4_OFF; }
  else                     { f = f5; HW = 49;   local = p - INV_L5_OFF; }
  int b   = local / (RT * HW);
  int rem = local - b * RT * HW;          // t*HW + y*W + x
  const float* base = f + (long)b * RC * RT * HW + rem;
  float ss = 0.f;
  for (int c = 0; c < RC; ++c) {
    float v = base[(long)c * RT * HW];
    ss += v * v;
  }
  invn[p] = 1.f / fmaxf(sqrtf(ss), 1e-12f);
}

// ---------------------------------------------------------------------------
// 2) ROI-align (+norm) -> pooled f16 [ROWS][C], and base_feat max -> d_out
// ---------------------------------------------------------------------------
__global__ __launch_bounds__(256) void pool_kernel(const float* __restrict__ f2,
                                                   const float* __restrict__ f3,
                                                   const float* __restrict__ f4,
                                                   const float* __restrict__ f5,
                                                   const float* __restrict__ rois,
                                                   const float* __restrict__ im_info,
                                                   const float* __restrict__ invn,
                                                   _Float16* __restrict__ pooled,
                                                   float* __restrict__ base_feat) {
  const int nt = blockIdx.x;             // 0..2047
  const int n  = nt >> 3, t = nt & 7;
  const int c  = threadIdx.x;
  const float* r = rois + n * 33;

  float hs = 0.f, wsz = 0.f;
  for (int tt = 0; tt < RT; ++tt) {
    hs  += r[4 + 4 * tt] - r[2 + 4 * tt] + 1.f;
    wsz += r[3 + 4 * tt] - r[1 + 4 * tt] + 1.f;
  }
  hs *= 0.125f; wsz *= 0.125f;
  float lv = rintf(logf(sqrtf(hs * wsz) / 112.f) + 4.f);
  lv = fminf(fmaxf(lv, 2.f), 5.f);
  int lvl = (int)lv;

  const float* f; int H, W; long invbase;
  if (lvl == 2)      { f = f2; H = 56; W = 56; invbase = INV_L2_OFF; }
  else if (lvl == 3) { f = f3; H = 28; W = 28; invbase = INV_L3_OFF; }
  else if (lvl == 4) { f = f4; H = 14; W = 14; invbase = INV_L4_OFF; }
  else               { f = f5; H = 7;  W = 7;  invbase = INV_L5_OFF; }
  const int HW = H * W;

  const float scale = (float)W / im_info[0];
  const int b = (int)r[0];
  float x1 = r[1 + 4 * t] * scale, y1 = r[2 + 4 * t] * scale;
  float x2 = r[3 + 4 * t] * scale, y2 = r[4 + 4 * t] * scale;
  float rw = fmaxf(x2 - x1, 1.f), rh = fmaxf(y2 - y1, 1.f);

  const float* fc = f + ((long)(b * RC + c) * RT + t) * HW;
  const float* iv = invn + invbase + (long)(b * RT + t) * HW;
  _Float16* prow = pooled + (long)nt * SAMP * RC + c;

  float mx = -3.4e38f;
  for (int s = 0; s < SAMP; ++s) {
    int py = s / 7, px = s - py * 7;
    float xs = x1 + rw * ((px + 0.5f) * (1.f / 7.f));
    float ys = y1 + rh * ((py + 0.5f) * (1.f / 7.f));
    float xf = floorf(xs), yf = floorf(ys);
    float fx = xs - xf,   fy = ys - yf;
    int x0  = min(max((int)xf, 0), W - 1);
    int x1i = min(max((int)xf + 1, 0), W - 1);
    int y0  = min(max((int)yf, 0), H - 1);
    int y1i = min(max((int)yf + 1, 0), H - 1);
    int i00 = y0 * W + x0,  i01 = y0 * W + x1i;
    int i10 = y1i * W + x0, i11 = y1i * W + x1i;
    float v00 = fc[i00] * iv[i00], v01 = fc[i01] * iv[i01];
    float v10 = fc[i10] * iv[i10], v11 = fc[i11] * iv[i11];
    float v = v00 * (1.f - fy) * (1.f - fx) + v01 * (1.f - fy) * fx
            + v10 * fy * (1.f - fx)        + v11 * fy * fx;
    mx = fmaxf(mx, v);
    prow[s * RC] = (_Float16)v;
  }
  base_feat[((long)n * RC + c) * RT + t] = mx;   // (n, c, t) layout
}

// ---------------------------------------------------------------------------
// 3) 1x1 conv GEMM: y[r,d] = sum_c pooled[r,c]*conv_w[d,c] + conv_b[d]
//    scattered into flat layout yT[n][d*392 + t*49 + s]  (f16)
// ---------------------------------------------------------------------------
__global__ __launch_bounds__(256) void conv_gemm(const _Float16* __restrict__ A,
                                                 const float* __restrict__ Wc,
                                                 const float* __restrict__ bc,
                                                 _Float16* __restrict__ Y) {
  __shared__ _Float16 lA[256 * LDS_PITCH];
  __shared__ _Float16 lB[64 * LDS_PITCH];
  const int tid = threadIdx.x, lane = tid & 31, wv = tid >> 5;
  const int rBase = blockIdx.x * 256;
  const int nb    = blockIdx.y * 64;
  v8f acc[2][4] = {};
  for (int kb = 0; kb < RC; kb += 32) {
#ifdef HAVE_TDM
    STAGE_A_TILE(lA, A + (long)rBase * RC + kb, RC, tid);
#else
    for (int j = 0; j < 4; ++j) {
      int q = tid + 256 * j; int row = q >> 2, ck = (q & 3) * 8;
      *(v8h*)&lA[row * LDS_PITCH + ck] =
          *(const v8h*)&A[(long)(rBase + row) * RC + kb + ck];
    }
#endif
    for (int j = 0; j < 2; ++j) {          // B: conv_w[d][c] native [N][K], b128 loads
      int q = tid + 256 * j; int dl = q >> 3, c4 = (q & 7) * 4;
      v4f v = *(const v4f*)&Wc[(long)(nb + dl) * RC + kb + c4];
      v4h h; h[0] = (_Float16)v[0]; h[1] = (_Float16)v[1];
             h[2] = (_Float16)v[2]; h[3] = (_Float16)v[3];
      *(v4h*)&lB[dl * LDS_PITCH + c4] = h;
    }
#ifdef HAVE_TDM
    __builtin_amdgcn_s_wait_tensorcnt(0);
#endif
    __syncthreads();
    v16h bf[4];
    for (int ntl = 0; ntl < 4; ++ntl) bf[ntl] = frag_b(lB, ntl * 16, lane);
    for (int mt = 0; mt < 2; ++mt) {
      v16h a = frag_a(lA, wv * 32 + mt * 16, lane);
      for (int ntl = 0; ntl < 4; ++ntl)
        acc[mt][ntl] = WMMA_F16(a, bf[ntl], acc[mt][ntl]);
    }
    __syncthreads();
  }
  for (int mt = 0; mt < 2; ++mt)
    for (int ntl = 0; ntl < 4; ++ntl) {
      int d = nb + ntl * 16 + (lane & 15);
      float bias = bc[d];
      for (int g = 0; g < 8; ++g) {
        int rr = rBase + wv * 32 + mt * 16 + g + ((lane & 16) ? 8 : 0);
        int n  = rr / 392, s = rr - n * 392;
        Y[(long)n * ROWS + (long)d * 392 + s] = (_Float16)(acc[mt][ntl][g] + bias);
      }
    }
}

// ---------------------------------------------------------------------------
// 4) Big GEMM: (256 x 100352) @ (100352 x 2048), K split 8x, per-split partials
// ---------------------------------------------------------------------------
__global__ __launch_bounds__(256) void big_gemm(const _Float16* __restrict__ A,
                                                const float* __restrict__ W1,
                                                float* __restrict__ H1p) {
  __shared__ _Float16 lA[256 * LDS_PITCH];
  __shared__ _Float16 lB[64 * LDS_PITCH];
  const int tid = threadIdx.x, lane = tid & 31, wv = tid >> 5;
  const int nb = blockIdx.x * 64;
  const int k0 = blockIdx.y * KCHUNK;
  v8f acc[2][4] = {};
  for (int kb = k0; kb < k0 + KCHUNK; kb += 32) {
    __builtin_prefetch(W1 + (long)(kb + 32) * D1 + nb, 0, 1);   // global_prefetch_b8
#ifdef HAVE_TDM
    STAGE_A_TILE(lA, A + kb, ROWS, tid);
#else
    for (int j = 0; j < 4; ++j) {
      int q = tid + 256 * j; int row = q >> 2, ck = (q & 3) * 8;
      *(v8h*)&lA[row * LDS_PITCH + ck] =
          *(const v8h*)&A[(long)row * ROWS + kb + ck];
    }
#endif
    {                                       // B: w1 32x64 -> [N][K], b128 loads
      int col4 = (tid & 15) * 4, k2 = (tid >> 4) * 2;
      const float* p0 = &W1[(long)(kb + k2) * D1 + nb + col4];
      v4f vA = *(const v4f*)p0;
      v4f vB = *(const v4f*)(p0 + D1);
      for (int e = 0; e < 4; ++e) {
        v2h h; h[0] = (_Float16)vA[e]; h[1] = (_Float16)vB[e];
        *(v2h*)&lB[(col4 + e) * LDS_PITCH + k2] = h;
      }
    }
#ifdef HAVE_TDM
    __builtin_amdgcn_s_wait_tensorcnt(0);
#endif
    __syncthreads();
    v16h bf[4];
    for (int ntl = 0; ntl < 4; ++ntl) bf[ntl] = frag_b(lB, ntl * 16, lane);
    for (int mt = 0; mt < 2; ++mt) {
      v16h a = frag_a(lA, wv * 32 + mt * 16, lane);
      for (int ntl = 0; ntl < 4; ++ntl)
        acc[mt][ntl] = WMMA_F16(a, bf[ntl], acc[mt][ntl]);
    }
    __syncthreads();
  }
  float* out = H1p + (long)blockIdx.y * RN * D1;
  for (int mt = 0; mt < 2; ++mt)
    for (int ntl = 0; ntl < 4; ++ntl) {
      int col = nb + ntl * 16 + (lane & 15);
      for (int g = 0; g < 8; ++g) {
        int row = wv * 32 + mt * 16 + g + ((lane & 16) ? 8 : 0);
        out[(long)row * D1 + col] = acc[mt][ntl][g];
      }
    }
}

__global__ __launch_bounds__(256) void reduce_h1(const float* __restrict__ H1p,
                                                 const float* __restrict__ b1,
                                                 float* __restrict__ H1) {
  int i = blockIdx.x * 256 + threadIdx.x;          // 0 .. RN*D1-1
  float s = b1[i & (D1 - 1)];
  for (int j = 0; j < KSPLIT; ++j) s += H1p[(long)j * RN * D1 + i];
  H1[i] = s;                                        // relu applied at next load
}

// ---------------------------------------------------------------------------
// 5) hdn2 = relu(relu(H1) @ w2 + b2) -> f16 [256][512]
// ---------------------------------------------------------------------------
__global__ __launch_bounds__(256) void mlp2_gemm(const float* __restrict__ H1,
                                                 const float* __restrict__ W2,
                                                 const float* __restrict__ b2,
                                                 _Float16* __restrict__ H2) {
  __shared__ _Float16 lA[256 * LDS_PITCH];
  __shared__ _Float16 lB[64 * LDS_PITCH];
  const int tid = threadIdx.x, lane = tid & 31, wv = tid >> 5;
  const int nb = blockIdx.x * 64;
  v8f acc[2][4] = {};
  for (int kb = 0; kb < D1; kb += 32) {
    for (int j = 0; j < 8; ++j) {           // A: relu(H1) f32->f16, b128 loads
      int q = tid + 256 * j; int row = q >> 3, k4 = (q & 7) * 4;
      v4f v = *(const v4f*)&H1[(long)row * D1 + kb + k4];
      v4h h; h[0] = (_Float16)fmaxf(v[0], 0.f); h[1] = (_Float16)fmaxf(v[1], 0.f);
             h[2] = (_Float16)fmaxf(v[2], 0.f); h[3] = (_Float16)fmaxf(v[3], 0.f);
      *(v4h*)&lA[row * LDS_PITCH + k4] = h;
    }
    {                                        // B: w2 32x64 -> [N][K], b128 loads
      int col4 = (tid & 15) * 4, k2 = (tid >> 4) * 2;
      const float* p0 = &W2[(long)(kb + k2) * D2 + nb + col4];
      v4f vA = *(const v4f*)p0;
      v4f vB = *(const v4f*)(p0 + D2);
      for (int e = 0; e < 4; ++e) {
        v2h h; h[0] = (_Float16)vA[e]; h[1] = (_Float16)vB[e];
        *(v2h*)&lB[(col4 + e) * LDS_PITCH + k2] = h;
      }
    }
    __syncthreads();
    v16h bf[4];
    for (int ntl = 0; ntl < 4; ++ntl) bf[ntl] = frag_b(lB, ntl * 16, lane);
    for (int mt = 0; mt < 2; ++mt) {
      v16h a = frag_a(lA, wv * 32 + mt * 16, lane);
      for (int ntl = 0; ntl < 4; ++ntl)
        acc[mt][ntl] = WMMA_F16(a, bf[ntl], acc[mt][ntl]);
    }
    __syncthreads();
  }
  for (int mt = 0; mt < 2; ++mt)
    for (int ntl = 0; ntl < 4; ++ntl) {
      int col = nb + ntl * 16 + (lane & 15);
      float bias = b2[col];
      for (int g = 0; g < 8; ++g) {
        int row = wv * 32 + mt * 16 + g + ((lane & 16) ? 8 : 0);
        H2[(long)row * D2 + col] = (_Float16)fmaxf(acc[mt][ntl][g] + bias, 0.f);
      }
    }
}

// ---------------------------------------------------------------------------
// 6) bbox = H2 @ wb + bb  (256x512 @ 512x32) -> d_out[0..8191]
// ---------------------------------------------------------------------------
__global__ __launch_bounds__(256) void bbox_gemm(const _Float16* __restrict__ H2,
                                                 const float* __restrict__ Wb,
                                                 const float* __restrict__ bb,
                                                 float* __restrict__ out) {
  __shared__ _Float16 lA[256 * LDS_PITCH];
  __shared__ _Float16 lB[32 * LDS_PITCH];
  const int tid = threadIdx.x, lane = tid & 31, wv = tid >> 5;
  v8f acc[2][2] = {};
  for (int kb = 0; kb < D2; kb += 32) {
#ifdef HAVE_TDM
    STAGE_A_TILE(lA, H2 + kb, D2, tid);
#else
    for (int j = 0; j < 4; ++j) {
      int q = tid + 256 * j; int row = q >> 2, ck = (q & 3) * 8;
      *(v8h*)&lA[row * LDS_PITCH + ck] =
          *(const v8h*)&H2[(long)row * D2 + kb + ck];
    }
#endif
    if (tid < 128) {                         // wb 32x32 -> [N][K], b128 loads
      int col4 = (tid & 7) * 4, k2 = (tid >> 3) * 2;
      const float* p0 = &Wb[(long)(kb + k2) * DOUT + col4];
      v4f vA = *(const v4f*)p0;
      v4f vB = *(const v4f*)(p0 + DOUT);
      for (int e = 0; e < 4; ++e) {
        v2h h; h[0] = (_Float16)vA[e]; h[1] = (_Float16)vB[e];
        *(v2h*)&lB[(col4 + e) * LDS_PITCH + k2] = h;
      }
    }
#ifdef HAVE_TDM
    __builtin_amdgcn_s_wait_tensorcnt(0);
#endif
    __syncthreads();
    v16h bf[2];
    for (int ntl = 0; ntl < 2; ++ntl) bf[ntl] = frag_b(lB, ntl * 16, lane);
    for (int mt = 0; mt < 2; ++mt) {
      v16h a = frag_a(lA, wv * 32 + mt * 16, lane);
      for (int ntl = 0; ntl < 2; ++ntl)
        acc[mt][ntl] = WMMA_F16(a, bf[ntl], acc[mt][ntl]);
    }
    __syncthreads();
  }
  for (int mt = 0; mt < 2; ++mt)
    for (int ntl = 0; ntl < 2; ++ntl) {
      int col = ntl * 16 + (lane & 15);
      float bias = bb[col];
      for (int g = 0; g < 8; ++g) {
        int row = wv * 32 + mt * 16 + g + ((lane & 16) ? 8 : 0);
        out[row * DOUT + col] = acc[mt][ntl][g] + bias;
      }
    }
}

// ---------------------------------------------------------------------------
extern "C" void kernel_launch(void* const* d_in, const int* in_sizes, int n_in,
                              void* d_out, int out_size, void* d_ws, size_t ws_size,
                              hipStream_t stream) {
  const float* feat2   = (const float*)d_in[0];
  const float* feat3   = (const float*)d_in[1];
  const float* feat4   = (const float*)d_in[2];
  const float* feat5   = (const float*)d_in[3];
  const float* rois    = (const float*)d_in[4];
  // d_in[5] = gt_rois (unused by the reference)
  const float* im_info = (const float*)d_in[6];
  const float* conv_w  = (const float*)d_in[7];
  const float* conv_b  = (const float*)d_in[8];
  const float* w1      = (const float*)d_in[9];
  const float* b1      = (const float*)d_in[10];
  const float* w2      = (const float*)d_in[11];
  const float* b2      = (const float*)d_in[12];
  const float* wb      = (const float*)d_in[13];
  const float* bbv     = (const float*)d_in[14];

  char* ws = (char*)d_ws;
  float*     invn   = (float*)(ws + O_INVN);
  _Float16*  pooled = (_Float16*)(ws + O_POOLED);
  _Float16*  yT     = (_Float16*)(ws + O_YT);
  float*     h1p    = (float*)(ws + O_H1P);
  float*     h1     = (float*)(ws + O_H1);
  _Float16*  h2     = (_Float16*)(ws + O_H2);
  float*     out    = (float*)d_out;

  norm_kernel<<<(INV_TOTAL + 255) / 256, 256, 0, stream>>>(feat2, feat3, feat4, feat5, invn);
  pool_kernel<<<RN * RT, 256, 0, stream>>>(feat2, feat3, feat4, feat5, rois, im_info,
                                           invn, pooled, out + RN * DOUT);
  conv_gemm<<<dim3(ROWS / 256, RC / 64), 256, 0, stream>>>(pooled, conv_w, conv_b, yT);
  big_gemm<<<dim3(D1 / 64, KSPLIT), 256, 0, stream>>>(yT, w1, h1p);
  reduce_h1<<<(RN * D1) / 256, 256, 0, stream>>>(h1p, b1, h1);
  mlp2_gemm<<<D2 / 64, 256, 0, stream>>>(h1, w2, b2, h2);
  bbox_gemm<<<1, 256, 0, stream>>>(h2, wb, bbv, out);
}
